// Attention_23046794510345
// MI455X (gfx1250) — matrix operand order repaired and checked by reading, not compiled, not run on hardware
//
#include <hip/hip_runtime.h>

// Flash attention for MI455X (gfx1250): BH=32, S=2048, D=128, fp32 I/O.
// bf16 WMMA (v_wmma_f32_16x16x32_bf16) for QK^T and PV, online softmax with
// v_permlane16_b32 lane reductions, software-pipelined LDS fragment loads.

#define BH 32
#define SEQ 2048
#define HD 128
#define BM 128          // Q rows per block (8 waves x 16 rows)
#define BN 64           // KV rows per tile
#define NWAVE 8
#define KSTRIDE 136     // bf16 elems per K row in LDS (128 + 8 pad; 272B = 17*16B)
#define VSTRIDE 72      // bf16 elems per Vt row (64 + 8 pad; 144B = 9*16B)
#define PSTRIDE 72      // bf16 elems per P row

typedef __attribute__((ext_vector_type(16))) __bf16 v16bf;
typedef __attribute__((ext_vector_type(8)))  __bf16 v8bf;
typedef __attribute__((ext_vector_type(8)))  float  v8f;
typedef __attribute__((ext_vector_type(4)))  float  v4f;

union Frag16 { v16bf v; v8bf h[2]; };
union PairBF { __bf16 b[2]; unsigned u; };

// ---- XOR-lane reductions within each 16-lane half-group (VALU pipe).
#if __has_builtin(__builtin_amdgcn_permlane16)
template <unsigned LO, unsigned HI>
__device__ inline float lane_xor(float x) {
    unsigned v = __builtin_bit_cast(unsigned, x);
    unsigned r = __builtin_amdgcn_permlane16(v, v, LO, HI, false, false);
    return __builtin_bit_cast(float, r);
}
__device__ inline float redmax16(float x) {
    x = fmaxf(x, lane_xor<0x67452301u, 0xEFCDAB89u>(x));  // xor 1
    x = fmaxf(x, lane_xor<0x45670123u, 0xCDEF89ABu>(x));  // xor 2
    x = fmaxf(x, lane_xor<0x01234567u, 0x89ABCDEFu>(x));  // xor 4
    x = fmaxf(x, lane_xor<0xFEDCBA98u, 0x76543210u>(x));  // xor 8
    return x;
}
__device__ inline float redsum16(float x) {
    x += lane_xor<0x67452301u, 0xEFCDAB89u>(x);
    x += lane_xor<0x45670123u, 0xCDEF89ABu>(x);
    x += lane_xor<0x01234567u, 0x89ABCDEFu>(x);
    x += lane_xor<0xFEDCBA98u, 0x76543210u>(x);
    return x;
}
#else
__device__ inline float redmax16(float x) {
    x = fmaxf(x, __shfl_xor(x, 1));
    x = fmaxf(x, __shfl_xor(x, 2));
    x = fmaxf(x, __shfl_xor(x, 4));
    x = fmaxf(x, __shfl_xor(x, 8));
    return x;
}
__device__ inline float redsum16(float x) {
    x += __shfl_xor(x, 1); x += __shfl_xor(x, 2);
    x += __shfl_xor(x, 4); x += __shfl_xor(x, 8);
    return x;
}
#endif

__device__ inline v8f wmma_bf16(v16bf a, v16bf b, v8f c) {
    return __builtin_amdgcn_wmma_f32_16x16x32_bf16(false, a, false, b,
                                                   (short)0, c, false, false);
}

__global__ __launch_bounds__(256)
void attn_fwd_kernel(const float* __restrict__ K,
                     const float* __restrict__ Q,
                     const float* __restrict__ V,
                     float* __restrict__ O) {
    __shared__ __bf16 lK[BN * KSTRIDE];            // 17408 B
    __shared__ __bf16 lVt[HD * VSTRIDE];           // 18432 B  (V transposed: [d][key])
    __shared__ __bf16 lP[NWAVE * 16 * PSTRIDE];    // 18432 B

    const int tid  = threadIdx.x;
    const int lane = tid & 31;
    const int wave = tid >> 5;
    const int row  = lane & 15;     // row (A frag) or column (B/C frag) within 16
    const int hi   = lane >> 4;     // lane half-group
    const int bh   = blockIdx.y;
    const int qbase = blockIdx.x * BM + wave * 16;

    const float scale = 0.08838834764831845f;      // 1/sqrt(128)

    // ---- Load Q once into 4 bf16 A-fragments (16x32 each), scale folded in.
    // A layout: low lanes hold K={0..7,16..23}, high lanes K={8..15,24..31}.
    v16bf qa[4];
    {
        const float* Qp = Q + ((size_t)bh * SEQ + qbase + row) * HD;
        #pragma unroll
        for (int kb = 0; kb < 4; ++kb) {
            const int k0 = kb * 32 + hi * 8;
            v4f x0 = *(const v4f*)(Qp + k0);
            v4f x1 = *(const v4f*)(Qp + k0 + 4);
            v4f y0 = *(const v4f*)(Qp + k0 + 16);
            v4f y1 = *(const v4f*)(Qp + k0 + 20);
            v16bf f;
            #pragma unroll
            for (int i = 0; i < 4; ++i) {
                f[i]      = (__bf16)(x0[i] * scale);
                f[4 + i]  = (__bf16)(x1[i] * scale);
                f[8 + i]  = (__bf16)(y0[i] * scale);
                f[12 + i] = (__bf16)(y1[i] * scale);
            }
            qa[kb] = f;
        }
    }

    // ---- Online-softmax state: VGPR r <-> query row (r + hi*8) of the wave tile.
    v8f o_acc[8];
    #pragma unroll
    for (int i = 0; i < 8; ++i) o_acc[i] = (v8f){0.f,0.f,0.f,0.f,0.f,0.f,0.f,0.f};
    float m_run[8], l_run[8];
    #pragma unroll
    for (int r = 0; r < 8; ++r) { m_run[r] = -INFINITY; l_run[r] = 0.f; }

    const size_t kv_base = (size_t)bh * SEQ * HD;
    // Staging coordinates (hoisted).
    const int kr = tid >> 2;             // K: key row 0..63
    const int kc = (tid & 3) * 32;       // K: 32-elem column chunk
    const int vkp = (tid & 31) * 2;      // V: even key of a key pair
    const int vd0 = (tid >> 5) * 16;     // V: 16 d-rows

    for (int nt = 0; nt < SEQ / BN; ++nt) {
        const int ntb = nt * BN;
        // ---- Stage K tile (row-major bf16) and V tile (transposed bf16).
        {
            const float* Kp = K + kv_base + (size_t)(ntb + kr) * HD + kc;
            #pragma unroll
            for (int i = 0; i < 4; ++i) {
                v4f a = ((const v4f*)Kp)[2 * i];
                v4f b = ((const v4f*)Kp)[2 * i + 1];
                v8bf f;
                #pragma unroll
                for (int j = 0; j < 4; ++j) {
                    f[j]     = (__bf16)a[j];
                    f[4 + j] = (__bf16)b[j];
                }
                *(v8bf*)&lK[kr * KSTRIDE + kc + 8 * i] = f;   // ds_store_b128
            }
            const float* Vp0 = V + kv_base + (size_t)(ntb + vkp) * HD + vd0;
            const float* Vp1 = Vp0 + HD;
            #pragma unroll
            for (int i = 0; i < 4; ++i) {
                v4f a = ((const v4f*)Vp0)[i];
                v4f b = ((const v4f*)Vp1)[i];
                #pragma unroll
                for (int j = 0; j < 4; ++j) {
                    PairBF pk;
                    pk.b[0] = (__bf16)a[j];          // key vkp
                    pk.b[1] = (__bf16)b[j];          // key vkp+1
                    *(unsigned*)&lVt[(vd0 + 4 * i + j) * VSTRIDE + vkp] = pk.u;
                }
            }
        }
        __syncthreads();

        // ---- Prefetch next KV tile into cache (global_prefetch_b8, no regs).
        if (nt + 1 < SEQ / BN) {
            const int nb = ntb + BN;
            __builtin_prefetch(K + kv_base + (size_t)(nb + kr) * HD + kc, 0, 0);
            __builtin_prefetch(V + kv_base + (size_t)(nb + kr) * HD + kc, 0, 0);
        }

        // ---- Scores: four 16x16 column tiles; 16 WMMAs, software-pipelined
        // B-fragment loads (ping-pong) so WMMAs don't drain the DS queue.
        v8f sc[4];
        #pragma unroll
        for (int ct = 0; ct < 4; ++ct) sc[ct] = (v8f){0.f,0.f,0.f,0.f,0.f,0.f,0.f,0.f};
        {
            Frag16 bb[2];
            const __bf16* kbase = &lK[row * KSTRIDE + hi * 8];
            // u = kb*4 + ct ; ct fastest keeps consecutive WMMAs on distinct accs.
            const __bf16* p0 = kbase;                       // kb=0, ct=0
            bb[0].h[0] = *(const v8bf*)p0;
            bb[0].h[1] = *(const v8bf*)(p0 + 16);
            #pragma unroll
            for (int u = 0; u < 16; ++u) {
                const int nu = u + 1;
                if (nu < 16) {
                    const __bf16* pn = kbase + (nu & 3) * (16 * KSTRIDE) + (nu >> 2) * 32;
                    bb[nu & 1].h[0] = *(const v8bf*)pn;
                    bb[nu & 1].h[1] = *(const v8bf*)(pn + 16);
                }
                sc[u & 3] = wmma_bf16(qa[u >> 2], bb[u & 1].v, sc[u & 3]);
            }
        }

        // ---- Online softmax: elementwise max/sum across 4 tiles, then one
        // 4-step permlane reduction per row (amortized over 64 keys).
        float alpha[8];
        #pragma unroll
        for (int r = 0; r < 8; ++r) {
            float x = fmaxf(fmaxf(sc[0][r], sc[1][r]), fmaxf(sc[2][r], sc[3][r]));
            x = redmax16(x);
            float mn = fmaxf(m_run[r], x);
            alpha[r] = __expf(m_run[r] - mn);
            m_run[r] = mn;
            float p0 = __expf(sc[0][r] - mn);
            float p1 = __expf(sc[1][r] - mn);
            float p2 = __expf(sc[2][r] - mn);
            float p3 = __expf(sc[3][r] - mn);
            sc[0][r] = p0; sc[1][r] = p1; sc[2][r] = p2; sc[3][r] = p3;
            float rs = redsum16((p0 + p1) + (p2 + p3));
            l_run[r] = l_run[r] * alpha[r] + rs;
        }
        #pragma unroll
        for (int db = 0; db < 8; ++db)
            #pragma unroll
            for (int r = 0; r < 8; ++r) o_acc[db][r] *= alpha[r];

        // ---- P (f32 C-layout) -> bf16 row-major in per-wave LDS.
        __bf16* pp = &lP[wave * 16 * PSTRIDE];
        #pragma unroll
        for (int r = 0; r < 8; ++r) {
            const int rr = r + hi * 8;
            pp[rr * PSTRIDE + row]      = (__bf16)sc[0][r];
            pp[rr * PSTRIDE + 16 + row] = (__bf16)sc[1][r];
            pp[rr * PSTRIDE + 32 + row] = (__bf16)sc[2][r];
            pp[rr * PSTRIDE + 48 + row] = (__bf16)sc[3][r];
        }
        __asm__ volatile("" ::: "memory");  // keep DS stores before DS reloads (HW in-order per wave)

        // ---- Reload P as two 16x32 A-fragments (keys 0-31, 32-63).
        Frag16 pa[2];
        const __bf16* prow = pp + row * PSTRIDE + hi * 8;
        pa[0].h[0] = *(const v8bf*)prow;
        pa[0].h[1] = *(const v8bf*)(prow + 16);
        pa[1].h[0] = *(const v8bf*)(prow + 32);
        pa[1].h[1] = *(const v8bf*)(prow + 48);

        // ---- O += P * V: key-chunk outer, d-block inner => consecutive WMMAs
        // hit 8 distinct accumulators; ping-pong V-fragment loads.
        {
            const __bf16* vbase = &lVt[row * VSTRIDE + hi * 8];
            Frag16 bv[2];
            bv[0].h[0] = *(const v8bf*)vbase;               // kc=0, db=0
            bv[0].h[1] = *(const v8bf*)(vbase + 16);
            #pragma unroll
            for (int u = 0; u < 16; ++u) {
                const int nu = u + 1;
                if (nu < 16) {
                    const __bf16* pn = vbase + (nu & 7) * (16 * VSTRIDE) + (nu >> 3) * 32;
                    bv[nu & 1].h[0] = *(const v8bf*)pn;
                    bv[nu & 1].h[1] = *(const v8bf*)(pn + 16);
                }
                o_acc[u & 7] = wmma_bf16(pa[u >> 3].v, bv[u & 1].v, o_acc[u & 7]);
            }
        }
        __syncthreads();
    }

    // ---- Epilogue: normalize and store fp32.
    float* Op = O + ((size_t)bh * SEQ + qbase) * HD;
    #pragma unroll
    for (int r = 0; r < 8; ++r) {
        const float inv = 1.0f / l_run[r];
        const int rr = r + hi * 8;
        #pragma unroll
        for (int db = 0; db < 8; ++db)
            Op[(size_t)rr * HD + db * 16 + row] = o_acc[db][r] * inv;
    }
}

extern "C" void kernel_launch(void* const* d_in, const int* in_sizes, int n_in,
                              void* d_out, int out_size, void* d_ws, size_t ws_size,
                              hipStream_t stream) {
    // setup_inputs() order: key, query, value (all fp32)
    const float* K = (const float*)d_in[0];
    const float* Q = (const float*)d_in[1];
    const float* V = (const float*)d_in[2];
    float* O = (float*)d_out;
    (void)in_sizes; (void)n_in; (void)out_size; (void)d_ws; (void)ws_size;

    dim3 grid(SEQ / BM, BH);   // 16 x 32
    dim3 block(256);
    attn_fwd_kernel<<<grid, block, 0, stream>>>(K, Q, V, O);
}
